// LuongAttnDecoderRNN_63350767616511
// MI455X (gfx1250) — matrix operand order repaired and checked by reading, not compiled, not run on hardware
//
#include <hip/hip_runtime.h>

#define B 32
#define S 64
#define T 64
#define H 1024
#define V 32000
#define KC 256

typedef __attribute__((ext_vector_type(16))) __bf16         v16bf;
typedef __attribute__((ext_vector_type(16))) unsigned short v16u;
typedef __attribute__((ext_vector_type(8)))  unsigned short v8u;
typedef __attribute__((ext_vector_type(8)))  float          v8f;
typedef __attribute__((ext_vector_type(4)))  float          v4f;
typedef int v4i __attribute__((vector_size(16)));   // matches builtin param type

#if __has_builtin(__builtin_amdgcn_global_load_async_to_lds_b128) && \
    __has_builtin(__builtin_amdgcn_s_wait_asynccnt)
#define HAVE_ASYNC_LDS 1
#endif

__device__ __forceinline__ unsigned short f2bf(float f) {
  unsigned int x = __float_as_uint(f);
  x += 0x7fffu + ((x >> 16) & 1u);          // round-to-nearest-even
  return (unsigned short)(x >> 16);
}
__device__ __forceinline__ float sigm(float x) { return 1.0f / (1.0f + __expf(-x)); }

// ---------------------------------------------------------------------------
// Generic WMMA GEMM:  C[M,N] = A[M,K](bf16, row-major) * Bw[N,K]^T (bf16) + bias
// 256 threads (8 waves). blockIdx.x -> 128-col N tile (16 per wave),
// blockIdx.y -> 32-row M tile (2 WMMA M-tiles per wave). A panel is staged
// through a double-buffered LDS panel; on toolchains exposing the gfx1250
// async-LDS path the staging overlaps with WMMA compute via ASYNCcnt.
// ---------------------------------------------------------------------------
__global__ __launch_bounds__(256)
void gemm_bf16(const unsigned short* __restrict__ A, int lda,
               const unsigned short* __restrict__ Bw, int ldb,
               const float* __restrict__ bias,
               float* __restrict__ C, int ldc, int K)
{
  __shared__ __align__(16) unsigned short As[2][32][KC + 8]; // +8 pad vs bank conflicts
  const int tid   = threadIdx.x;
  const int wave  = tid >> 5;
  const int lane  = tid & 31;
  const int nl    = lane & 15;       // N within tile (B/C/D), M within tile (A)
  const int khalf = lane >> 4;       // lane group 0/1
  const size_t mbase = (size_t)blockIdx.y * 32;
  const int    nbase = blockIdx.x * 128 + wave * 16;

  // cooperative A-panel staging: 32 rows x KC cols per chunk, 64B per thread
  const int arow = tid >> 3;         // 0..31
  const int acol = (tid & 7) * 32;   // 0,32,...,224
  const unsigned short* Arow = A + (mbase + arow) * (size_t)lda + acol;
  const unsigned short* Bn   = Bw + (size_t)(nbase + nl) * (size_t)ldb + khalf * 16;

  auto stage = [&](int kc, int buf) {
    const unsigned short* src = Arow + kc;
    unsigned short* dst = &As[buf][arow][acol];
#ifdef HAVE_ASYNC_LDS
    __attribute__((address_space(1))) v4i* gp =
        (__attribute__((address_space(1))) v4i*)(unsigned long long)src;
    __attribute__((address_space(3))) v4i* lp =
        (__attribute__((address_space(3))) v4i*)(unsigned)(unsigned long long)dst;
#pragma unroll
    for (int i = 0; i < 4; ++i)
      __builtin_amdgcn_global_load_async_to_lds_b128(gp + i, lp + i, 0, 0);
#else
#pragma unroll
    for (int i = 0; i < 4; ++i)
      *(v8u*)(dst + i * 8) = *(const v8u*)(src + i * 8);
#endif
  };

  v8f acc0 = {}; v8f acc1 = {};
  const int am0 = nl;                // A rows 0..15 of the 32-row panel
  const int am1 = nl + 16;           // A rows 16..31

  stage(0, 0);
#ifdef HAVE_ASYNC_LDS
  __builtin_amdgcn_s_wait_asynccnt(0);
#endif
  __syncthreads();

  int cur = 0;
  for (int kc = 0; kc < K; kc += KC) {
    if (kc + KC < K) stage(kc + KC, cur ^ 1);       // overlap next panel with compute
    __builtin_prefetch(Bn + kc + KC, 0, 3);          // WGP-scope prefetch of B stream
#pragma unroll
    for (int ks = 0; ks < KC; ks += 32) {
      // B fragment (32x16): lanes 0-15 hold K=0..15 (N=lane), lanes 16-31 K=16..31
      v16u bu = *(const v16u*)(Bn + kc + ks);
      // A fragments (16x32): lane<16 -> K 0-15; lane>=16 -> K 16-31
      v16u au0, au1;
      ((v8u*)&au0)[0] = *(const v8u*)&As[cur][am0][ks + khalf * 16];
      ((v8u*)&au0)[1] = *(const v8u*)&As[cur][am0][ks + khalf * 16 + 8];
      ((v8u*)&au1)[0] = *(const v8u*)&As[cur][am1][ks + khalf * 16];
      ((v8u*)&au1)[1] = *(const v8u*)&As[cur][am1][ks + khalf * 16 + 8];
      v16bf bb = __builtin_bit_cast(v16bf, bu);
      acc0 = __builtin_amdgcn_wmma_f32_16x16x32_bf16(false, __builtin_bit_cast(v16bf, au0),
                                                     false, bb, (short)0, acc0, false, false);
      acc1 = __builtin_amdgcn_wmma_f32_16x16x32_bf16(false, __builtin_bit_cast(v16bf, au1),
                                                     false, bb, (short)0, acc1, false, false);
    }
#ifdef HAVE_ASYNC_LDS
    __builtin_amdgcn_s_wait_asynccnt(0);             // next panel landed in LDS
#endif
    __syncthreads();
    cur ^= 1;
  }

  const float bv = bias ? bias[nbase + nl] : 0.0f;
  const int   m0 = khalf * 8;        // D layout: VGPR r -> M = r + 8*khalf, N = nl
#pragma unroll
  for (int r = 0; r < 8; ++r) {
    C[(mbase +      m0 + r) * (size_t)ldc + (nbase + nl)] = acc0[r] + bv;
    C[(mbase + 16 + m0 + r) * (size_t)ldc + (nbase + nl)] = acc1[r] + bv;
  }
}

// ---------------------------------------------------------------------------
// fp32 -> bf16 conversion, 8 elements/thread (b128 in / b128 out)
// ---------------------------------------------------------------------------
__global__ __launch_bounds__(256)
void conv_bf16_k(const float* __restrict__ s, unsigned short* __restrict__ d, long n)
{
  long i = ((long)blockIdx.x * 256 + threadIdx.x) * 8;
  if (i + 8 <= n) {
    v4f a = *(const v4f*)(s + i);
    v4f b = *(const v4f*)(s + i + 4);
    v8u o;
#pragma unroll
    for (int j = 0; j < 4; ++j) { o[j] = f2bf(a[j]); o[4 + j] = f2bf(b[j]); }
    *(v8u*)(d + i) = o;
  } else {
    for (long k = i; k < n; ++k) d[k] = f2bf(s[k]);
  }
}

// teacher-forced input gather: row r = b*T + t holds embedding[inputs[b][t]]
__global__ __launch_bounds__(256)
void embed_gather_k(const float* __restrict__ emb, const int* __restrict__ tgt,
                    unsigned short* __restrict__ out)
{
  long i = ((long)blockIdx.x * 256 + threadIdx.x) * 8;
  if (i >= (long)B * T * H) return;
  int  h = (int)(i % H);
  long r = i / H;
  int  t = (int)(r % T);
  int  b = (int)(r / T);
  int idx = (t == 0) ? 0 : tgt[b * T + t - 1];       // SOS = 0
  const float* src = emb + (long)idx * H + h;
  v4f a = *(const v4f*)src;
  v4f c = *(const v4f*)(src + 4);
  v8u o;
#pragma unroll
  for (int j = 0; j < 4; ++j) { o[j] = f2bf(a[j]); o[4 + j] = f2bf(c[j]); }
  *(v8u*)(out + i) = o;
}

__global__ __launch_bounds__(256)
void init_h_k(const float* __restrict__ eh, float* __restrict__ h,
              unsigned short* __restrict__ hbf)
{
  int i = blockIdx.x * 256 + threadIdx.x;
  if (i < B * H) { float v = eh[i]; h[i] = v; hbf[i] = f2bf(v); }
}

// ---------------------------------------------------------------------------
// Fused GRU gates + Luong attention + concat.  One block per batch element.
// ---------------------------------------------------------------------------
__global__ __launch_bounds__(256)
void gru_attn_k(const float* __restrict__ GI, const float* __restrict__ GH,
                const float* __restrict__ b_hh, const float* __restrict__ keys,
                const float* __restrict__ enc, float* __restrict__ h,
                unsigned short* __restrict__ h_bf, unsigned short* __restrict__ concat_bf,
                float* __restrict__ out_attn, float* __restrict__ out_hfinal, int t)
{
  __shared__ float hn_s[H];
  __shared__ float sc[S];
  __shared__ float al[S];
  const int b = blockIdx.x, tid = threadIdx.x;
  const float* gi = GI + ((long)b * T + t) * (3 * H);
  const float* gh = GH + (long)b * (3 * H);
#pragma unroll
  for (int jj = 0; jj < H / 256; ++jj) {
    int   j = tid + jj * 256;
    float r = sigm(gi[j] + gh[j] + b_hh[j]);
    float z = sigm(gi[H + j] + gh[H + j] + b_hh[H + j]);
    float n = tanhf(gi[2 * H + j] + r * (gh[2 * H + j] + b_hh[2 * H + j]));
    float hv = (1.0f - z) * n + z * h[(long)b * H + j];
    hn_s[j] = hv;
    h[(long)b * H + j] = hv;                       // per-element in-place: safe
    unsigned short hb = f2bf(hv);
    h_bf[(long)b * H + j] = hb;
    concat_bf[(long)b * 2 * H + j] = hb;
    if (t == T - 1) out_hfinal[(long)b * H + j] = hv;
  }
  __syncthreads();
  // scores[s] = dot(h_new, keys[b,s,:]) — 4 lanes per s
  {
    const int s = tid >> 2, l4 = tid & 3;
    const float* kk = keys + ((long)b * S + s) * H;
    float acc = 0.0f;
    for (int k = l4; k < H; k += 4) acc += hn_s[k] * kk[k];
    acc += __shfl_xor(acc, 1, 32);
    acc += __shfl_xor(acc, 2, 32);
    if (l4 == 0) sc[s] = acc;
  }
  __syncthreads();
  // softmax over S=64 (redundant per-thread reduction — S is tiny)
  float m = -1e30f;
  for (int i = 0; i < S; ++i) m = fmaxf(m, sc[i]);
  float sum = 0.0f;
  for (int i = 0; i < S; ++i) sum += __expf(sc[i] - m);
  float inv = 1.0f / sum;
  if (tid < S) {
    float a = __expf(sc[tid] - m) * inv;
    al[tid] = a;
    out_attn[((long)b * T + t) * S + tid] = a;
  }
  __syncthreads();
  // context[j] = sum_s attn[s] * enc[b,s,j]; write bf16 into second half of concat
#pragma unroll
  for (int jj = 0; jj < H / 256; ++jj) {
    int j = tid + jj * 256;
    float c = 0.0f;
    for (int i = 0; i < S; ++i) c += al[i] * enc[((long)b * S + i) * H + j];
    concat_bf[(long)b * 2 * H + H + j] = f2bf(c);
  }
}

// ---------------------------------------------------------------------------
// log_softmax over V per batch row; writes log_probs[b, t, :]
// ---------------------------------------------------------------------------
__global__ __launch_bounds__(256)
void log_softmax_k(const float* __restrict__ logits, float* __restrict__ out, int t)
{
  __shared__ float red[256];
  const int b = blockIdx.x, tid = threadIdx.x;
  const float* row = logits + (long)b * V;
  float m = -1e30f;
  for (int v = tid; v < V; v += 256) m = fmaxf(m, row[v]);
  red[tid] = m; __syncthreads();
  for (int s = 128; s > 0; s >>= 1) { if (tid < s) red[tid] = fmaxf(red[tid], red[tid + s]); __syncthreads(); }
  m = red[0]; __syncthreads();
  float sum = 0.0f;
  for (int v = tid; v < V; v += 256) sum += __expf(row[v] - m);
  red[tid] = sum; __syncthreads();
  for (int s = 128; s > 0; s >>= 1) { if (tid < s) red[tid] += red[tid + s]; __syncthreads(); }
  const float lse = m + __logf(red[0]);
  float* o = out + ((long)b * T + t) * V;
  for (int v = tid; v < V; v += 256) o[v] = row[v] - lse;
}

// ---------------------------------------------------------------------------
extern "C" void kernel_launch(void* const* d_in, const int* in_sizes, int n_in,
                              void* d_out, int out_size, void* d_ws, size_t ws_size,
                              hipStream_t stream)
{
  (void)in_sizes; (void)n_in; (void)out_size; (void)ws_size;
  const float* enc   = (const float*)d_in[0];
  const float* ehid  = (const float*)d_in[1];
  const int*   tgt   = (const int*)  d_in[2];
  const float* emb   = (const float*)d_in[3];
  const float* W_ih  = (const float*)d_in[4];
  const float* W_hh  = (const float*)d_in[5];
  const float* b_ih  = (const float*)d_in[6];
  const float* b_hh  = (const float*)d_in[7];
  const float* Wa    = (const float*)d_in[8];
  const float* ba    = (const float*)d_in[9];
  const float* W_out = (const float*)d_in[10];
  const float* b_out = (const float*)d_in[11];

  float* out_lp   = (float*)d_out;                          // [B,T,V]
  float* out_hfin = out_lp + (size_t)B * T * V;             // [B,H]
  float* out_attn = out_hfin + (size_t)B * H;               // [B,T,S]

  // workspace carve-out (256B aligned)
  char*  ws  = (char*)d_ws;
  size_t off = 0;
  auto carve = [&](size_t bytes) -> char* {
    char* p = ws + off;
    off += (bytes + 255) & ~(size_t)255;
    return p;
  };
  unsigned short* Wout_bf = (unsigned short*)carve((size_t)V * 2 * H * 2); // 128 MB: fits L2
  unsigned short* Wih_bf  = (unsigned short*)carve((size_t)3 * H * H * 2);
  unsigned short* Whh_bf  = (unsigned short*)carve((size_t)3 * H * H * 2);
  unsigned short* Wa_bf   = (unsigned short*)carve((size_t)H * H * 2);
  unsigned short* enc_bf  = (unsigned short*)carve((size_t)B * S * H * 2);
  unsigned short* emb_bf  = (unsigned short*)carve((size_t)B * T * H * 2);
  float*          GI      = (float*)carve((size_t)B * T * 3 * H * 4);
  float*          keys    = (float*)carve((size_t)B * S * H * 4);
  float*          GH      = (float*)carve((size_t)B * 3 * H * 4);
  float*          hbuf    = (float*)carve((size_t)B * H * 4);
  unsigned short* h_bf    = (unsigned short*)carve((size_t)B * H * 2);
  unsigned short* cat_bf  = (unsigned short*)carve((size_t)B * 2 * H * 2);
  float*          logits  = (float*)carve((size_t)B * V * 4);

  auto cblk8 = [](long n) { return (unsigned)((n / 8 + 255) / 256); };

  // --- one-time precompute (also replayed in the timed graph) ---
  conv_bf16_k<<<cblk8((long)V * 2 * H), 256, 0, stream>>>(W_out, Wout_bf, (long)V * 2 * H);
  conv_bf16_k<<<cblk8((long)3 * H * H), 256, 0, stream>>>(W_ih, Wih_bf, (long)3 * H * H);
  conv_bf16_k<<<cblk8((long)3 * H * H), 256, 0, stream>>>(W_hh, Whh_bf, (long)3 * H * H);
  conv_bf16_k<<<cblk8((long)H * H),     256, 0, stream>>>(Wa, Wa_bf, (long)H * H);
  conv_bf16_k<<<cblk8((long)B * S * H), 256, 0, stream>>>(enc, enc_bf, (long)B * S * H);
  embed_gather_k<<<cblk8((long)B * T * H), 256, 0, stream>>>(emb, tgt, emb_bf);
  init_h_k<<<(B * H + 255) / 256, 256, 0, stream>>>(ehid, hbuf, h_bf);

  // keys = enc @ Wa^T + ba : [B*S, H] x [H, H]
  gemm_bf16<<<dim3(H / 128, (B * S) / 32), 256, 0, stream>>>(
      enc_bf, H, Wa_bf, H, ba, keys, H, H);
  // GI = emb @ W_ih^T + b_ih : [B*T, H] x [H, 3H]  (hoisted out of the t-loop)
  gemm_bf16<<<dim3((3 * H) / 128, (B * T) / 32), 256, 0, stream>>>(
      emb_bf, H, Wih_bf, H, b_ih, GI, 3 * H, H);

  // --- sequential decode loop ---
  for (int t = 0; t < T; ++t) {
    // GH = h @ W_hh^T : [32, 1024] x [1024, 3072]
    gemm_bf16<<<dim3((3 * H) / 128, 1), 256, 0, stream>>>(
        h_bf, H, Whh_bf, H, nullptr, GH, 3 * H, H);
    // gates + attention + concat
    gru_attn_k<<<B, 256, 0, stream>>>(GI, GH, b_hh, keys, enc, hbuf,
                                      h_bf, cat_bf, out_attn, out_hfin, t);
    // logits = [h | ctx] @ W_out^T + b_out : [32, 2048] x [2048, 32000]
    gemm_bf16<<<dim3(V / 128, 1), 256, 0, stream>>>(
        cat_bf, 2 * H, Wout_bf, 2 * H, b_out, logits, V, 2 * H);
    // log_softmax over V -> log_probs[:, t, :]
    log_softmax_k<<<B, 256, 0, stream>>>(logits, out_lp, t);
  }
}
